// VariateMambaBlock_8400956031272
// MI455X (gfx1250) — matrix operand order repaired and checked by reading, not compiled, not run on hardware
//
#include <hip/hip_runtime.h>
#include <hip/hip_bf16.h>

// ---------------------------------------------------------------------------
// Mamba block forward for MI455X (gfx1250, wave32, WMMA).
// Large GEMMs: v_wmma_f32_16x16x32_bf16 with B-panel staged in LDS via the
// CDNA5 async DMA path (global_load_async_to_lds_b128 / ASYNCcnt), double
// buffered. Small GEMM (N=80) uses a register-only single-wave path.
// ---------------------------------------------------------------------------

#define D_MODEL 768
#define D_INNER 1536
#define D_CONV  4
#define D_STATE 16
#define DT_RANK 48
#define DT_PAD  64     // DT_RANK padded to a multiple of 32 for WMMA K-loop
#define BB      2
#define LL      2048
#define BL      (BB * LL)   // 4096 rows

typedef __bf16 bf16_t;
typedef __attribute__((ext_vector_type(8)))  __bf16 v8bf;
typedef __attribute__((ext_vector_type(16))) __bf16 v16bf;
typedef __attribute__((ext_vector_type(8)))  float  v8f;

// ---------------------------------------------------------------------------
// CDNA5 async global->LDS DMA (16 bytes per lane, tracked by ASYNCcnt).
// lds_off is the 32-bit workgroup-relative LDS byte offset (low 32 bits of a
// generic pointer into __shared__, per ISA 10.2 aperture layout).
// ---------------------------------------------------------------------------
__device__ __forceinline__ void async_copy_b128(unsigned lds_off, const bf16_t* g) {
  asm volatile("global_load_async_to_lds_b128 %0, %1, off"
               :: "v"(lds_off), "v"((unsigned long long)g) : "memory");
}
__device__ __forceinline__ void wait_asynccnt_0() {
  asm volatile("s_wait_asynccnt 0x0" ::: "memory");
}
__device__ __forceinline__ void wait_asynccnt_1() {
  asm volatile("s_wait_asynccnt 0x1" ::: "memory");
}

// ---------------------------------------------------------------------------
// Fragment loaders (contiguous k order k = 16h + s, same for A and B, wave32).
// A (16x32 MxK): lane m = lane&15; kb=0 (lanes 0-15) / kb=16 (lanes 16-31);
//   VGPR0..7 hold K=kb+0..15 contiguous.
// B (32x16 KxN) from row-major W[N,K]: lane n = lane&15; lanes 0-15 hold
//   K = 0..15 contiguous, lanes 16-31 hold K = 16..31 contiguous.
// ---------------------------------------------------------------------------
__device__ __forceinline__ v16bf load_a_frag(const bf16_t* __restrict__ A, int lda,
                                             int m0, int k0, int lane) {
  const int m  = m0 + (lane & 15);
  const int kb = k0 + ((lane & 16) ? 16 : 0);
  const bf16_t* p = A + (size_t)m * lda + kb;
  union { v16bf v; struct { v8bf lo; v8bf hi; } s; } f;
  f.s.lo = *(const v8bf*)(p);
  f.s.hi = *(const v8bf*)(p + 8);
  return f.v;
}

__device__ __forceinline__ v16bf load_b_frag_g(const bf16_t* __restrict__ W, int ldb,
                                               int n0, int k0, int lane) {
  const int n  = n0 + (lane & 15);
  const int kb = k0 + ((lane & 16) ? 16 : 0);
  const bf16_t* p = W + (size_t)n * ldb + kb;
  union { v16bf v; struct { v8bf lo; v8bf hi; } s; } f;
  f.s.lo = *(const v8bf*)(p);
  f.s.hi = *(const v8bf*)(p + 8);
  return f.v;
}

// B fragment from an LDS tile laid out [64 n][32 k] bf16, k contiguous.
__device__ __forceinline__ v16bf load_b_frag_lds(const bf16_t* tile, int n_loc, int lane) {
  const int kb = (lane & 16) ? 16 : 0;
  const bf16_t* p = tile + (n_loc + (lane & 15)) * 32 + kb;
  union { v16bf v; struct { v8bf lo; v8bf hi; } s; } f;
  f.s.lo = *(const v8bf*)(p);
  f.s.hi = *(const v8bf*)(p + 8);
  return f.v;
}

// ---------------------------------------------------------------------------
// LDS-staged WMMA GEMM: C[M,N] = A[M,K] * W[N,K]^T  (bf16 in, f32 out).
// 256 threads = 8 waves; block tile M=128 x N=64; wave w does rows m0+16w.
// B slice (64x32) is async-DMA'd to LDS, double buffered across the K loop.
// EPI: 0=store, 1=+bias then softplus, 2=+residual.
// ---------------------------------------------------------------------------
template <int EPI>
__global__ __launch_bounds__(256)
void wmma_gemm_lds_kernel(const bf16_t* __restrict__ A,
                          const bf16_t* __restrict__ W,
                          float* __restrict__ C,
                          int K, int lda, int ldb, int ldc,
                          const float* __restrict__ aux, int ldaux) {
  __shared__ bf16_t smem[2][64 * 32];
  const int tid  = threadIdx.x;
  const int lane = tid & 31;
  const int wid  = tid >> 5;                 // 0..7
  const int m0   = blockIdx.y * 128 + wid * 16;
  const int n0b  = blockIdx.x * 64;

  // Staging: lane e moves 16B; row j = e/4 of the 64x32 B slice, k chunk c.
  const int j = tid >> 2;                    // 0..63
  const int c = (tid & 3) * 8;               // 0,8,16,24
  const unsigned lds0 = (unsigned)(size_t)&smem[0][j * 32 + c];
  const unsigned lds1 = (unsigned)(size_t)&smem[1][j * 32 + c];
  const bf16_t* gsrc  = W + (size_t)(n0b + j) * ldb + c;

  const int nsteps = K / 32;
  async_copy_b128(lds0, gsrc);               // prologue: stage k-step 0

  v8f acc[4];
  v8f zero = {};
#pragma unroll
  for (int s = 0; s < 4; ++s) acc[s] = zero;

  for (int t = 0; t < nsteps; ++t) {
    const int cur = t & 1;
    if (t + 1 < nsteps) {                    // stage next slice into other buf
      async_copy_b128(cur ? lds0 : lds1, gsrc + (size_t)(t + 1) * 32);
      wait_asynccnt_1();                     // current slice resident
    } else {
      wait_asynccnt_0();
    }
    __syncthreads();

    v16bf af = load_a_frag(A, lda, m0, t * 32, lane);
    const bf16_t* tile = &smem[cur][0];
#pragma unroll
    for (int s = 0; s < 4; ++s) {
      v16bf bfr = load_b_frag_lds(tile, s * 16, lane);
      acc[s] = __builtin_amdgcn_wmma_f32_16x16x32_bf16(
          false, af, false, bfr, (short)0, acc[s], false, false);
    }
    __syncthreads();                         // readers done before next overwrite
  }

  // C/D layout: VGPR i -> row m0 + i (+8 for lanes 16-31), col = lane&15.
  const int row_base = m0 + ((lane & 16) ? 8 : 0);
  const int col_in   = lane & 15;
#pragma unroll
  for (int s = 0; s < 4; ++s) {
    const int col = n0b + 16 * s + col_in;
#pragma unroll
    for (int i = 0; i < 8; ++i) {
      const int row = row_base + i;
      float v = acc[s][i];
      if (EPI == 1) {                        // dt_proj: + bias, softplus
        v += aux[col];
        v = (v > 20.f) ? v : __logf(1.f + __expf(v));
      } else if (EPI == 2) {                 // out_proj: + residual
        v += aux[(size_t)row * ldaux + col];
      }
      C[(size_t)row * ldc + col] = v;
    }
  }
}

// ---------------------------------------------------------------------------
// Register-only single-wave GEMM for the small-N x_proj (N=80).
// ---------------------------------------------------------------------------
__global__ void wmma_gemm_small_kernel(const bf16_t* __restrict__ A,
                                       const bf16_t* __restrict__ W,
                                       float* __restrict__ C,
                                       int K, int lda, int ldb, int ldc) {
  const int lane = threadIdx.x & 31;
  const int m0 = blockIdx.y * 16;
  const int n0 = blockIdx.x * 16;
  v8f acc = {};
  for (int k0 = 0; k0 < K; k0 += 32) {
    v16bf af = load_a_frag(A, lda, m0, k0, lane);
    v16bf bfr = load_b_frag_g(W, ldb, n0, k0, lane);
    acc = __builtin_amdgcn_wmma_f32_16x16x32_bf16(
        false, af, false, bfr, (short)0, acc, false, false);
  }
  const int row_base = m0 + ((lane & 16) ? 8 : 0);
  const int col = n0 + (lane & 15);
#pragma unroll
  for (int i = 0; i < 8; ++i)
    C[(size_t)(row_base + i) * ldc + col] = acc[i];
}

// ---------------------------------------------------------------------------
// LayerNorm over D_MODEL, writes bf16 operand for in_proj GEMM.
// ---------------------------------------------------------------------------
__global__ void ln_kernel(const float* __restrict__ x,
                          const float* __restrict__ w,
                          const float* __restrict__ b,
                          bf16_t* __restrict__ h) {
  const int row = blockIdx.x;
  const float* xr = x + (size_t)row * D_MODEL;
  float s = 0.f, ss = 0.f;
  for (int c = threadIdx.x; c < D_MODEL; c += 256) {
    float v = xr[c];
    s += v; ss += v * v;
  }
#pragma unroll
  for (int off = 16; off > 0; off >>= 1) {
    s  += __shfl_down(s,  off, 32);
    ss += __shfl_down(ss, off, 32);
  }
  __shared__ float red_s[8], red_q[8];
  const int wid = threadIdx.x >> 5;
  if ((threadIdx.x & 31) == 0) { red_s[wid] = s; red_q[wid] = ss; }
  __syncthreads();
  float sum = 0.f, sumsq = 0.f;
#pragma unroll
  for (int i = 0; i < 8; ++i) { sum += red_s[i]; sumsq += red_q[i]; }
  const float mu  = sum * (1.f / D_MODEL);
  const float var = sumsq * (1.f / D_MODEL) - mu * mu;
  const float rs  = rsqrtf(var + 1e-5f);
  for (int c = threadIdx.x; c < D_MODEL; c += 256)
    h[(size_t)row * D_MODEL + c] = (bf16_t)((xr[c] - mu) * rs * w[c] + b[c]);
}

// ---------------------------------------------------------------------------
// f32 -> bf16 conversion with optional right zero-padding of columns.
// ---------------------------------------------------------------------------
__global__ void cvt_pad_kernel(const float* __restrict__ src, int src_ld, int scols,
                               bf16_t* __restrict__ dst, int dcols, int total) {
  const int e = blockIdx.x * blockDim.x + threadIdx.x;
  if (e >= total) return;
  const int r = e / dcols, c = e - r * dcols;
  dst[e] = (c < scols) ? (bf16_t)src[(size_t)r * src_ld + c] : (bf16_t)0.f;
}

// ---------------------------------------------------------------------------
// Depthwise causal conv1d (k=4) + bias + SiLU. Reads the u-half of xr.
// ---------------------------------------------------------------------------
__global__ void conv_silu_kernel(const float* __restrict__ xr,
                                 const float* __restrict__ cw,
                                 const float* __restrict__ cb,
                                 float* __restrict__ u2f,
                                 bf16_t* __restrict__ u2b) {
  const int e = blockIdx.x * blockDim.x + threadIdx.x;
  if (e >= BL * D_INNER) return;
  const int d = e % D_INNER;
  const int m = e / D_INNER;          // m = b*L + l
  const int l = m % LL;
  float acc = cb[d];
#pragma unroll
  for (int j = 0; j < D_CONV; ++j) {
    const int ll = l - (D_CONV - 1) + j;
    if (ll >= 0)
      acc += xr[(size_t)(m - (D_CONV - 1) + j) * (2 * D_INNER) + d] * cw[d * D_CONV + j];
  }
  const float sv = acc / (1.f + __expf(-acc));
  u2f[e] = sv;
  u2b[e] = (bf16_t)sv;
}

// ---------------------------------------------------------------------------
// Selective scan: one thread per (b,d), 16-state vector in registers, serial
// over L. Fuses y = scan + u*D, then g = y * silu(res) -> bf16 for out_proj.
// ---------------------------------------------------------------------------
__global__ void scan_kernel(const float* __restrict__ delta,
                            const float* __restrict__ u2,
                            const float* __restrict__ xdbl,
                            const float* __restrict__ A_log,
                            const float* __restrict__ Dv,
                            const float* __restrict__ xr,
                            bf16_t* __restrict__ g) {
  const int d = blockIdx.x * blockDim.x + threadIdx.x;  // 0..D_INNER-1
  const int b = blockIdx.y;
  float Arow[D_STATE];
#pragma unroll
  for (int n = 0; n < D_STATE; ++n) Arow[n] = -__expf(A_log[d * D_STATE + n]);
  const float Dd = Dv[d];
  float st[D_STATE];
#pragma unroll
  for (int n = 0; n < D_STATE; ++n) st[n] = 0.f;

  for (int l = 0; l < LL; ++l) {
    const int m = b * LL + l;
    const float dlt = delta[(size_t)m * D_INNER + d];
    const float uu  = u2[(size_t)m * D_INNER + d];
    const float du  = dlt * uu;
    const float4* bc = (const float4*)(xdbl + (size_t)m * (DT_RANK + 2 * D_STATE) + DT_RANK);
    const float4 B0 = bc[0], B1 = bc[1], B2 = bc[2], B3 = bc[3];
    const float4 C0 = bc[4], C1 = bc[5], C2 = bc[6], C3 = bc[7];
    const float Bf[D_STATE] = {B0.x, B0.y, B0.z, B0.w, B1.x, B1.y, B1.z, B1.w,
                               B2.x, B2.y, B2.z, B2.w, B3.x, B3.y, B3.z, B3.w};
    const float Cf[D_STATE] = {C0.x, C0.y, C0.z, C0.w, C1.x, C1.y, C1.z, C1.w,
                               C2.x, C2.y, C2.z, C2.w, C3.x, C3.y, C3.z, C3.w};
    float y = 0.f;
#pragma unroll
    for (int n = 0; n < D_STATE; ++n) {
      st[n] = __expf(dlt * Arow[n]) * st[n] + du * Bf[n];
      y += st[n] * Cf[n];
    }
    y += uu * Dd;
    const float r = xr[(size_t)m * (2 * D_INNER) + D_INNER + d];
    const float gr = y * (r / (1.f + __expf(-r)));
    g[(size_t)m * D_INNER + d] = (bf16_t)gr;
  }
}

// ---------------------------------------------------------------------------
extern "C" void kernel_launch(void* const* d_in, const int* in_sizes, int n_in,
                              void* d_out, int out_size, void* d_ws, size_t ws_size,
                              hipStream_t stream) {
  const float* x        = (const float*)d_in[0];   // [2,2048,768]
  const float* ln_w     = (const float*)d_in[1];
  const float* ln_b     = (const float*)d_in[2];
  const float* in_w     = (const float*)d_in[3];   // [3072,768]
  const float* conv_w   = (const float*)d_in[4];   // [1536,4]
  const float* conv_b   = (const float*)d_in[5];
  const float* xp_w     = (const float*)d_in[6];   // [80,1536]
  const float* dt_w     = (const float*)d_in[7];   // [1536,48]
  const float* dt_b     = (const float*)d_in[8];
  const float* A_log    = (const float*)d_in[9];   // [1536,16]
  const float* Dvec     = (const float*)d_in[10];
  const float* out_w    = (const float*)d_in[11];  // [768,1536]
  float* out            = (float*)d_out;

  char* ws = (char*)d_ws;
  size_t off = 0;
  auto alloc = [&](size_t bytes) -> void* {
    void* p = ws + off;
    off = (off + bytes + 255) & ~size_t(255);
    return p;
  };
  bf16_t* h_b    = (bf16_t*)alloc((size_t)BL * D_MODEL * 2);
  bf16_t* Win_b  = (bf16_t*)alloc((size_t)(2 * D_INNER) * D_MODEL * 2);
  bf16_t* Wxp_b  = (bf16_t*)alloc((size_t)(DT_RANK + 2 * D_STATE) * D_INNER * 2);
  bf16_t* Wdt_b  = (bf16_t*)alloc((size_t)D_INNER * DT_PAD * 2);
  bf16_t* Wout_b = (bf16_t*)alloc((size_t)D_MODEL * D_INNER * 2);
  float*  xr     = (float*) alloc((size_t)BL * (2 * D_INNER) * 4);
  float*  u2f    = (float*) alloc((size_t)BL * D_INNER * 4);
  bf16_t* u2b    = (bf16_t*)alloc((size_t)BL * D_INNER * 2);
  float*  xdbl   = (float*) alloc((size_t)BL * (DT_RANK + 2 * D_STATE) * 4);
  bf16_t* dtb    = (bf16_t*)alloc((size_t)BL * DT_PAD * 2);
  float*  dlt    = (float*) alloc((size_t)BL * D_INNER * 4);
  bf16_t* gb     = (bf16_t*)alloc((size_t)BL * D_INNER * 2);
  (void)ws_size; (void)in_sizes; (void)n_in; (void)out_size;

  // 1) LayerNorm -> bf16
  ln_kernel<<<BL, 256, 0, stream>>>(x, ln_w, ln_b, h_b);

  // 2) Weight conversions (f32 -> bf16, dt_proj K padded 48->64)
  {
    int t;
    t = 2 * D_INNER * D_MODEL;
    cvt_pad_kernel<<<(t + 255) / 256, 256, 0, stream>>>(in_w, D_MODEL, D_MODEL, Win_b, D_MODEL, t);
    t = (DT_RANK + 2 * D_STATE) * D_INNER;
    cvt_pad_kernel<<<(t + 255) / 256, 256, 0, stream>>>(xp_w, D_INNER, D_INNER, Wxp_b, D_INNER, t);
    t = D_INNER * DT_PAD;
    cvt_pad_kernel<<<(t + 255) / 256, 256, 0, stream>>>(dt_w, DT_RANK, DT_RANK, Wdt_b, DT_PAD, t);
    t = D_MODEL * D_INNER;
    cvt_pad_kernel<<<(t + 255) / 256, 256, 0, stream>>>(out_w, D_INNER, D_INNER, Wout_b, D_INNER, t);
  }

  // 3) in_proj: xr[4096,3072] = h[4096,768] @ in_w^T   (LDS/async staged)
  wmma_gemm_lds_kernel<0><<<dim3((2 * D_INNER) / 64, BL / 128), 256, 0, stream>>>(
      h_b, Win_b, xr, D_MODEL, D_MODEL, D_MODEL, 2 * D_INNER, nullptr, 0);

  // 4) depthwise conv + SiLU
  {
    int t = BL * D_INNER;
    conv_silu_kernel<<<(t + 255) / 256, 256, 0, stream>>>(xr, conv_w, conv_b, u2f, u2b);
  }

  // 5) x_proj: xdbl[4096,80] = u @ xp_w^T   (N=80 -> small kernel)
  wmma_gemm_small_kernel<<<dim3((DT_RANK + 2 * D_STATE) / 16, BL / 16), 32, 0, stream>>>(
      u2b, Wxp_b, xdbl, D_INNER, D_INNER, D_INNER, DT_RANK + 2 * D_STATE);

  // 6) dt slice -> bf16 padded [4096,64]
  {
    int t = BL * DT_PAD;
    cvt_pad_kernel<<<(t + 255) / 256, 256, 0, stream>>>(
        xdbl, DT_RANK + 2 * D_STATE, DT_RANK, dtb, DT_PAD, t);
  }

  // 7) dt_proj + bias + softplus: delta[4096,1536]
  wmma_gemm_lds_kernel<1><<<dim3(D_INNER / 64, BL / 128), 256, 0, stream>>>(
      dtb, Wdt_b, dlt, DT_PAD, DT_PAD, DT_PAD, D_INNER, dt_b, 0);

  // 8) selective scan + gate -> gb (bf16)
  scan_kernel<<<dim3(D_INNER / 256, BB), 256, 0, stream>>>(
      dlt, u2f, xdbl, A_log, Dvec, xr, gb);

  // 9) out_proj + residual -> d_out (f32)
  wmma_gemm_lds_kernel<2><<<dim3(D_MODEL / 64, BL / 128), 256, 0, stream>>>(
      gb, Wout_b, out, D_INNER, D_INNER, D_INNER, D_MODEL, x, D_MODEL);
}
